// SpikingGestureNet_46772193853474
// MI455X (gfx1250) — matrix operand order repaired and checked by reading, not compiled, not run on hardware
//
#include <hip/hip_runtime.h>

#define BATCH   8192
#define IN_F    128
#define HID     512
#define OUT_F   64
#define TSTEPS  64
#define MTILE   32
#define BETA_C  0.9048374180359595f   // exp(-1/10)

// Padded LDS row strides (in halfs): +8 halfs (16B) per row so consecutive
// matrix rows land 272B / 1040B apart -> per-lane bank rotation of 4 banks,
// conflict-free ds_load_b128 fragment loads (64 banks x 4B).
#define W1S  (IN_F + 8)    // 136
#define W2S  (HID + 8)     // 520
#define S1S  (HID + 8)     // 520
#define SPS  (IN_F + 8)    // 136

typedef _Float16 v16h __attribute__((ext_vector_type(16)));
typedef _Float16 v8h  __attribute__((ext_vector_type(8)));
typedef float    v8f  __attribute__((ext_vector_type(8)));

__device__ __forceinline__ v8f zero8() {
    v8f z;
#pragma unroll
    for (int i = 0; i < 8; ++i) z[i] = 0.0f;
    return z;
}

// A-fragment (16x32 f16, MxK) from row-major LDS [row][k], row stride `stride` halfs.
// Lanes 0-15: M=lane; lanes 16-31: M=lane-16, K offset +8.
// VGPRs 0-3 = K[hi*8..hi*8+7], VGPRs 4-7 = K[16+hi*8..16+hi*8+7] -> two aligned 16B loads.
__device__ __forceinline__ v16h ldsA(const _Float16* base, int row0, int stride, int k0, int lane) {
    const int m  = lane & 15;
    const int hi = lane >> 4;
    const _Float16* p = base + (row0 + m) * stride + k0 + hi * 8;
    v8h lo = *(const v8h*)p;
    v8h hh = *(const v8h*)(p + 16);
    return __builtin_shufflevector(lo, hh, 0,1,2,3,4,5,6,7,8,9,10,11,12,13,14,15);
}

// B-fragment (32x16 f16, KxN) with B[k][n] = W[n][k], W row-major, row stride kS halfs.
// Lanes 0-15: N=lane, K=0..7,16..23; lanes 16-31: N=lane-16, K=8..15,24..31 (matches ldsA).
__device__ __forceinline__ v16h ldsB(const _Float16* W, int n0, int kS, int k0, int lane) {
    const int n  = lane & 15;
    const int hi = lane >> 4;
    const _Float16* p = W + (n0 + n) * kS + k0 + hi * 8;
    v8h lo = *(const v8h*)p;
    v8h hh = *(const v8h*)(p + 16);
    return __builtin_shufflevector(lo, hh, 0,1,2,3,4,5,6,7,8,9,10,11,12,13,14,15);
}

__device__ __forceinline__ v8f wmma_f16(v16h a, v16h b, v8f c) {
    return __builtin_amdgcn_wmma_f32_16x16x32_f16(false, a, false, b, (short)0, c, false, false);
}

__global__ __launch_bounds__(256, 1)
void snn_lif2_kernel(const float* __restrict__ x, const float* __restrict__ noise,
                     const float* __restrict__ W1, const float* __restrict__ b1,
                     const float* __restrict__ W2, const float* __restrict__ b2,
                     float* __restrict__ out)
{
    extern __shared__ char smem_raw[];
    _Float16* sW1  = (_Float16*)smem_raw;          // 512 rows x W1S
    _Float16* sW2  = sW1 + HID * W1S;              // 64  rows x W2S
    _Float16* sS1  = sW2 + OUT_F * W2S;            // 32  rows x S1S
    _Float16* sSpk = sS1 + MTILE * S1S;            // 32  rows x SPS
    float*    sB1  = (float*)(sSpk + MTILE * SPS); // 512 f32
    float*    sB2  = sB1 + HID;                    // 64  f32

    const int tid     = threadIdx.x;
    const int lane    = tid & 31;
    const int w       = tid >> 5;            // wave id 0..7 (wave32)
    const int rowBase = blockIdx.x * MTILE;

    // ---- one-time setup: f16 weights + biases into padded LDS ----
#pragma unroll 4
    for (int i = tid; i < HID * IN_F; i += 256) {
        const int r = i >> 7, c = i & (IN_F - 1);
        sW1[r * W1S + c] = (_Float16)W1[i];
    }
#pragma unroll 4
    for (int i = tid; i < OUT_F * HID; i += 256) {
        const int r = i >> 9, c = i & (HID - 1);
        sW2[r * W2S + c] = (_Float16)W2[i];
    }
    for (int i = tid; i < HID; i += 256) sB1[i] = b1[i];
    if (tid < OUT_F) sB2[tid] = b2[tid];

    // sigmoid(10x): 16 consecutive elements per thread, kept in VGPRs
    float sig[16];
    {
        const float* xp = x + (size_t)rowBase * IN_F + tid * 16;
#pragma unroll
        for (int j = 0; j < 16; ++j) sig[j] = 1.0f / (1.0f + __expf(-10.0f * xp[j]));
    }

    // Layer-1: wave owns BOTH row tiles x 4 hidden col tiles -> each W1 B-fragment
    // is reused for 2 WMMAs (halves W1 LDS traffic).
    const int ctBase = w * 4;         // 4 col tiles per wave, 8 waves cover 32
    // Layer-2: wave owns one 16x16 output tile
    const int rt2 = w >> 2;
    const int ot  = w & 3;
    const int n   = lane & 15;
    const int hi  = lane >> 4;

    v8f mem1[8];                      // [rt*4 + c]
#pragma unroll
    for (int j = 0; j < 8; ++j) mem1[j] = zero8();
    v8f mem2 = zero8();
    v8f accv = zero8();

    __syncthreads();
    const float bias2 = sB2[ot * 16 + n];
    float bias1[4];
#pragma unroll
    for (int c = 0; c < 4; ++c) bias1[c] = sB1[(ctBase + c) * 16 + n];

    const int  spkRow = tid >> 3;             // 8 threads per 128-wide row
    const int  spkCol = (tid & 7) * 16;
    const float* noiseBase = noise + (size_t)rowBase * IN_F + (size_t)tid * 16;

    for (int t = 0; t < TSTEPS; ++t) {
        __syncthreads();   // prior-step sSpk/sS1 consumers done

        // ---- spike encode: spk = (noise < sigmoid(10x)) as f16 {0,1} ----
        const float* np_ = noiseBase + (size_t)t * ((size_t)BATCH * IN_F);
        _Float16* sp_ = sSpk + spkRow * SPS + spkCol;
#pragma unroll
        for (int j = 0; j < 16; j += 4) {
            float4 nv = *(const float4*)(np_ + j);
            sp_[j + 0] = (nv.x < sig[j + 0]) ? (_Float16)1.0f : (_Float16)0.0f;
            sp_[j + 1] = (nv.y < sig[j + 1]) ? (_Float16)1.0f : (_Float16)0.0f;
            sp_[j + 2] = (nv.z < sig[j + 2]) ? (_Float16)1.0f : (_Float16)0.0f;
            sp_[j + 3] = (nv.w < sig[j + 3]) ? (_Float16)1.0f : (_Float16)0.0f;
        }
        if (t + 1 < TSTEPS)
            __builtin_prefetch(np_ + (size_t)BATCH * IN_F, 0, 0);  // next step's noise
        __syncthreads();

        // ---- layer 1: mem1 = beta*mem1 + spk @ W1^T + b1 ; spike & reset ----
        v16h a1[2][4];
#pragma unroll
        for (int rt = 0; rt < 2; ++rt)
#pragma unroll
            for (int kt = 0; kt < 4; ++kt)
                a1[rt][kt] = ldsA(sSpk, rt * 16, SPS, kt * 32, lane);

#pragma unroll
        for (int c = 0; c < 4; ++c) {
            const int col0 = (ctBase + c) * 16;
            v16h bf[4];
#pragma unroll
            for (int kt = 0; kt < 4; ++kt)
                bf[kt] = ldsB(sW1, col0, W1S, kt * 32, lane);

#pragma unroll
            for (int rt = 0; rt < 2; ++rt) {
                v8f cc;
#pragma unroll
                for (int i = 0; i < 8; ++i) cc[i] = BETA_C * mem1[rt * 4 + c][i] + bias1[c];
#pragma unroll
                for (int kt = 0; kt < 4; ++kt)
                    cc = wmma_f16(a1[rt][kt], bf[kt], cc);
#pragma unroll
                for (int i = 0; i < 8; ++i) {
                    const float v = cc[i];
                    const bool sp = (v >= 1.0f);
                    mem1[rt * 4 + c][i] = sp ? 0.0f : v;              // reset-to-zero
                    sS1[(rt * 16 + i + 8 * hi) * S1S + col0 + n] =
                        sp ? (_Float16)1.0f : (_Float16)0.0f;
                }
            }
        }
        __syncthreads();

        // ---- layer 2: mem2 = beta*mem2 + s1 @ W2^T + b2 ; spike, reset, count ----
        v8f c2;
#pragma unroll
        for (int i = 0; i < 8; ++i) c2[i] = BETA_C * mem2[i] + bias2;
#pragma unroll
        for (int kt = 0; kt < 16; ++kt) {
            v16h a2 = ldsA(sS1, rt2 * 16, S1S, kt * 32, lane);
            v16h bf = ldsB(sW2, ot * 16, W2S, kt * 32, lane);
            c2 = wmma_f16(a2, bf, c2);
        }
#pragma unroll
        for (int i = 0; i < 8; ++i) {
            const float v = c2[i];
            const bool sp = (v >= 1.0f);
            mem2[i] = sp ? 0.0f : v;
            accv[i] += sp ? 1.0f : 0.0f;
        }
    }

    // ---- write spike counts: out[rowBase + rt2*16 + i + 8*hi][ot*16 + n] ----
#pragma unroll
    for (int i = 0; i < 8; ++i) {
        const size_t row = (size_t)(rowBase + rt2 * 16 + i + 8 * hi);
        out[row * OUT_F + ot * 16 + n] = accv[i];
    }
}

extern "C" void kernel_launch(void* const* d_in, const int* in_sizes, int n_in,
                              void* d_out, int out_size, void* d_ws, size_t ws_size,
                              hipStream_t stream) {
    const float* x     = (const float*)d_in[0];
    const float* noise = (const float*)d_in[1];
    const float* W1    = (const float*)d_in[2];
    const float* b1    = (const float*)d_in[3];
    const float* W2    = (const float*)d_in[4];
    const float* b2    = (const float*)d_in[5];
    float* out = (float*)d_out;

    const size_t smem =
        (size_t)(HID * W1S + OUT_F * W2S + MTILE * S1S + MTILE * SPS) * sizeof(_Float16)
        + (size_t)(HID + OUT_F) * sizeof(float);   // ~244 KB of the WGP's 320 KB LDS

    dim3 grid(BATCH / MTILE);   // 256 blocks
    dim3 block(256);            // 8 wave32 waves
    hipLaunchKernelGGL(snn_lif2_kernel, grid, block, smem, stream,
                       x, noise, W1, b1, W2, b2, out);
}
